// SimpleGRU_17033840296037
// MI455X (gfx1250) — compile-verified, matrix operand order checked by reading, not run on hardware
//
#include <hip/hip_runtime.h>

typedef __attribute__((ext_vector_type(16))) __bf16 v16bf;
typedef __attribute__((ext_vector_type(8)))  float  v8f;

#define T_IN    144
#define T_ALL   432
#define T_OUT   288
#define HID     256
#define G3      768
#define DIN     10
#define MROWS   16    // batch rows per workgroup
#define NTHR    512   // 16 waves

// Global-address-space pointer type: forces GLOBAL_LOAD_* (LOADcnt only)
// instead of FLAT_LOAD_* (LOADcnt + DScnt, ties up the LDS path).
typedef __attribute__((address_space(1))) const v16bf GFrag;

// K-index of element e (0..15) for half (lane>=16) in a 16-bit A/B fragment
// (CDNA5 ISA 7.12.2 "16-bit A-Matrix 16x32" table; B symmetric with N<->lane).
__device__ __host__ __forceinline__ int kmap(int e, int half) {
  int j = e >> 1, p = e & 1;
  return ((j < 4) ? 0 : 16) + half * 8 + ((j & 3) << 1) + p;
}

__device__ __forceinline__ v8f wmma_bf16(v16bf a, v16bf b, v8f c) {
  return __builtin_amdgcn_wmma_f32_16x16x32_bf16(
      false, a, false, b, (short)0, c, false, false);
}

// Opaque pointer: blocks LICM/CSE of weight loads across time steps, keeps
// them as per-step global loads served by the (weight-resident) L2.
__device__ __forceinline__ GFrag* opaque(const v16bf* p) {
  asm volatile("" : "+s"(p));
  return (GFrag*)p;
}

__device__ __forceinline__ float sigm(float x) {
  return 1.0f / (1.0f + __expf(-x));
}
__device__ __forceinline__ float tanh_(float x) {
  return 1.0f - 2.0f / (__expf(2.0f * x) + 1.0f);
}

// Convert a (768 x 256) fp32 row-major matrix into bf16 WMMA B-fragment order:
// dst[((ntile*8+kt)*32 + lane)*16 + e] = W[ntile*16 + lane%16][kt*32 + kmap(e, lane/16)]
__global__ void frag_convert(const float* __restrict__ W, unsigned short* __restrict__ dst) {
  int idx   = blockIdx.x * blockDim.x + threadIdx.x;   // 0 .. 196607
  int e     = idx & 15;
  int lane  = (idx >> 4) & 31;
  int f     = idx >> 9;
  int kt    = f & 7;
  int ntile = f >> 3;
  int row   = ntile * 16 + (lane & 15);
  int col   = kt * 32 + kmap(e, lane >> 4);
  __bf16 b  = (__bf16)W[row * HID + col];
  dst[idx]  = __builtin_bit_cast(unsigned short, b);
}

// Convert w_ih_0 (768 x 10) into zero-padded (K=32) bf16 B-fragments:
// dst[(ntile*32 + lane)*16 + e] = (k<10) ? W[ntile*16 + lane%16][k] : 0,  k = kmap(e, lane/16)
__global__ void frag_convert_ih0(const float* __restrict__ W, unsigned short* __restrict__ dst) {
  int idx   = blockIdx.x * blockDim.x + threadIdx.x;   // 0 .. 24575
  int e     = idx & 15;
  int lane  = (idx >> 4) & 31;
  int ntile = idx >> 9;
  int row   = ntile * 16 + (lane & 15);
  int col   = kmap(e, lane >> 4);
  float v   = (col < DIN) ? W[row * DIN + col] : 0.0f;
  __bf16 b  = (__bf16)v;
  dst[idx]  = __builtin_bit_cast(unsigned short, b);
}

__launch_bounds__(NTHR, 1)
__global__ void gru_fused(
    const float* __restrict__ input,   // (256,144,10)
    const float* __restrict__ bih0, const float* __restrict__ bhh0,
    const float* __restrict__ bih1, const float* __restrict__ bhh1,
    const float* __restrict__ wproj, const float* __restrict__ bproj,
    const unsigned short* __restrict__ wih0_u,   // padded B-frags (K=32)
    const unsigned short* __restrict__ whh0_u,
    const unsigned short* __restrict__ wih1_u,
    const unsigned short* __restrict__ whh1_u,
    float* __restrict__ out)           // (256,288)
{
  __shared__ __align__(64) __bf16 hA1[2][4096];   // h1, A-fragment order, double-buffered
  __shared__ __align__(64) __bf16 h2A[2][4096];   // h2, A-fragment order, double-buffered
  __shared__ __align__(64) __bf16 xA[512];        // x_t as padded 16x32 A-fragment
  __shared__ float projAcc[MROWS];

  const int tid  = threadIdx.x;
  const int w    = tid >> 5;        // wave = H-tile index 0..15
  const int lane = tid & 31;
  const int n16  = lane & 15;
  const int half = lane >> 4;
  const int b0   = blockIdx.x * MROWS;

  const v16bf* wih0f = (const v16bf*)wih0_u;
  const v16bf* whh0f = (const v16bf*)whh0_u;
  const v16bf* wih1f = (const v16bf*)wih1_u;
  const v16bf* whh1f = (const v16bf*)whh1_u;

  {
    __bf16* p1 = &hA1[0][0];
    __bf16* p2 = &h2A[0][0];
    for (int i = tid; i < 8192; i += NTHR) { p1[i] = (__bf16)0.0f; p2[i] = (__bf16)0.0f; }
  }

  // x-staging indices for this thread (one fragment element per thread).
  const int xlane = tid >> 4;              // fragment lane 0..31
  const int xe    = tid & 15;              // element 0..15
  const int xm    = xlane & 15;            // batch row
  const int xk    = kmap(xe, xlane >> 4);  // K index 0..31 (valid if <10)

  // Per-lane constants (single H-tile per wave).
  const int c = w * 16 + n16;                 // hidden column 0..255
  const float bR0  = bih0[c]       + bhh0[c];
  const float bZ0  = bih0[256 + c] + bhh0[256 + c];
  const float bXN0 = bih0[512 + c];
  const float bHN0 = bhh0[512 + c];
  const float bR1  = bih1[c]       + bhh1[c];
  const float bZ1  = bih1[256 + c] + bhh1[256 + c];
  const float bXN1 = bih1[512 + c];
  const float bHN1 = bhh1[512 + c];
  const float wpj  = wproj[c];
  const float bpj  = bproj[0];

  // Inverse kmap: LDS scatter base for h[m][c] in A-fragment order (+ i*16 per row).
  int stBase;
  {
    int kp = c & 31;
    int tt = kp >> 1, p = kp & 1;
    int half_a = (tt >> 2) & 1;
    int j  = ((tt >> 3) << 2) + (tt & 3);
    int e  = (j << 1) + p;
    stBase = (c >> 5) * 512 + (16 * half_a + 8 * half) * 16 + e;
  }

  float h1r[8] = {};
  float h2r[8] = {};
  __syncthreads();

  for (int t = 0; t < T_ALL; ++t) {
    const int rb = t & 1, wb = rb ^ 1;

    // Fresh (opaque, global-AS) weight bases each step.
    GFrag* i0t = opaque(wih0f);
    GFrag* w0t = opaque(whh0f);
    GFrag* i1t = opaque(wih1f);
    GFrag* h1t = opaque(whh1f);

    if (tid < MROWS) projAcc[tid] = 0.0f;
    {
      float v = 0.0f;
      if (t < T_IN && xk < DIN)
        v = input[(b0 + xm) * (T_IN * DIN) + t * DIN + xk];
      xA[tid] = (__bf16)v;
    }
    __syncthreads();

    // ---------------- layer 0 ----------------
    v8f accR, accZ, accXN, accHN;
    #pragma unroll
    for (int i = 0; i < 8; ++i) {
      accR[i]  = bR0;
      accZ[i]  = bZ0;
      accXN[i] = bXN0;
      accHN[i] = bHN0;
    }

    {
      // input side: one K=32 slice (x padded)
      const v16bf* xAP = (const v16bf*)&xA[0];
      v16bf ax  = xAP[lane];
      v16bf ibr = i0t[(w)      * 32 + lane];
      v16bf ibz = i0t[(16 + w) * 32 + lane];
      v16bf ibn = i0t[(32 + w) * 32 + lane];
      accR  = wmma_bf16(ax, ibr, accR);
      accZ  = wmma_bf16(ax, ibz, accZ);
      accXN = wmma_bf16(ax, ibn, accXN);

      // hidden side: K=256
      const v16bf* aP = (const v16bf*)&hA1[rb][0];
      #pragma unroll
      for (int kt = 0; kt < 8; ++kt) {
        v16bf a  = aP[kt * 32 + lane];
        v16bf br = w0t[((w)      * 8 + kt) * 32 + lane];
        v16bf bz = w0t[((16 + w) * 8 + kt) * 32 + lane];
        v16bf bn = w0t[((32 + w) * 8 + kt) * 32 + lane];
        accR  = wmma_bf16(a, br, accR);
        accZ  = wmma_bf16(a, bz, accZ);
        accHN = wmma_bf16(a, bn, accHN);
      }
    }

    #pragma unroll
    for (int i = 0; i < 8; ++i) {
      float r  = sigm(accR[i]);
      float z  = sigm(accZ[i]);
      float n  = tanh_(accXN[i] + r * accHN[i]);
      float hv = (1.0f - z) * n + z * h1r[i];
      h1r[i] = hv;
      hA1[wb][stBase + i * 16] = (__bf16)hv;
    }
    __syncthreads();   // h1(t) visible to all waves

    // ---------------- layer 1 ----------------
    v8f acc1R, acc1Z, acc1XN, acc1HN;
    #pragma unroll
    for (int i = 0; i < 8; ++i) {
      acc1R[i]  = bR1;
      acc1Z[i]  = bZ1;
      acc1XN[i] = bXN1;
      acc1HN[i] = bHN1;
    }

    {
      const v16bf* a1P = (const v16bf*)&hA1[wb][0];   // h1(t)
      const v16bf* a2P = (const v16bf*)&h2A[rb][0];   // h2(t-1)
      #pragma unroll
      for (int kt = 0; kt < 8; ++kt) {
        v16bf a1  = a1P[kt * 32 + lane];
        v16bf a2  = a2P[kt * 32 + lane];
        v16bf ir  = i1t[((w)      * 8 + kt) * 32 + lane];
        v16bf iz  = i1t[((16 + w) * 8 + kt) * 32 + lane];
        v16bf inn = i1t[((32 + w) * 8 + kt) * 32 + lane];
        v16bf hr  = h1t[((w)      * 8 + kt) * 32 + lane];
        v16bf hz  = h1t[((16 + w) * 8 + kt) * 32 + lane];
        v16bf hn  = h1t[((32 + w) * 8 + kt) * 32 + lane];
        // a1-group then a2-group: dependent acc pairs separated
        acc1R  = wmma_bf16(a1, ir,  acc1R);
        acc1Z  = wmma_bf16(a1, iz,  acc1Z);
        acc1XN = wmma_bf16(a1, inn, acc1XN);
        acc1R  = wmma_bf16(a2, hr,  acc1R);
        acc1Z  = wmma_bf16(a2, hz,  acc1Z);
        acc1HN = wmma_bf16(a2, hn,  acc1HN);
      }
    }

    float pm[8];
    #pragma unroll
    for (int i = 0; i < 8; ++i) {
      float r  = sigm(acc1R[i]);
      float z  = sigm(acc1Z[i]);
      float n  = tanh_(acc1XN[i] + r * acc1HN[i]);
      float hv = (1.0f - z) * n + z * h2r[i];
      h2r[i] = hv;
      h2A[wb][stBase + i * 16] = (__bf16)hv;
      pm[i] = hv * wpj;
    }

    if (t >= T_IN) {
      #pragma unroll
      for (int mask = 1; mask < 16; mask <<= 1)
        #pragma unroll
        for (int i = 0; i < 8; ++i)
          pm[i] += __shfl_xor(pm[i], mask, 32);
      if (n16 == 0) {
        #pragma unroll
        for (int i = 0; i < 8; ++i)
          atomicAdd(&projAcc[half * 8 + i], pm[i]);
      }
    }
    __syncthreads();
    if (t >= T_IN && tid < MROWS)
      out[(b0 + tid) * T_OUT + (t - T_IN)] = projAcc[tid] + bpj;
  }
}

extern "C" void kernel_launch(void* const* d_in, const int* in_sizes, int n_in,
                              void* d_out, int out_size, void* d_ws, size_t ws_size,
                              hipStream_t stream) {
  (void)in_sizes; (void)n_in; (void)out_size; (void)ws_size;
  const float* input = (const float*)d_in[0];
  const float* wih0  = (const float*)d_in[1];
  const float* whh0  = (const float*)d_in[2];
  const float* bih0  = (const float*)d_in[3];
  const float* bhh0  = (const float*)d_in[4];
  const float* wih1  = (const float*)d_in[5];
  const float* whh1  = (const float*)d_in[6];
  const float* bih1  = (const float*)d_in[7];
  const float* bhh1  = (const float*)d_in[8];
  const float* wproj = (const float*)d_in[9];
  const float* bproj = (const float*)d_in[10];
  float* out = (float*)d_out;

  const int NW  = G3 * HID;                // 196608 elements per recurrent matrix
  const int NW0 = G3 * 32;                 // 24576 elements for padded w_ih_0
  unsigned short* ws    = (unsigned short*)d_ws;
  unsigned short* whh0f = ws;
  unsigned short* wih1f = ws + NW;
  unsigned short* whh1f = ws + 2 * NW;
  unsigned short* wih0f = ws + 3 * NW;

  frag_convert<<<NW / 256, 256, 0, stream>>>(whh0, whh0f);
  frag_convert<<<NW / 256, 256, 0, stream>>>(wih1, wih1f);
  frag_convert<<<NW / 256, 256, 0, stream>>>(whh1, whh1f);
  frag_convert_ih0<<<NW0 / 256, 256, 0, stream>>>(wih0, wih0f);

  gru_fused<<<16, NTHR, 0, stream>>>(input, bih0, bhh0, bih1, bhh1,
                                     wproj, bproj, wih0f, whh0f, wih1f, whh1f, out);
}
